// Attention_68547678044697
// MI455X (gfx1250) — matrix operand order repaired and checked by reading, not compiled, not run on hardware
//
#include <hip/hip_runtime.h>

// ---------------------------------------------------------------------------
// Attention (channel-attention variant) for B=16, N=4096, C=384, H=8, D=48
// Heavy GEMMs: v_wmma_f32_16x16x32_bf16, weight tiles staged to LDS via the
// Tensor Data Mover (tensor_load_to_lds + s_wait_tensorcnt) on CDNA5/gfx1250.
// ---------------------------------------------------------------------------

typedef __bf16 bf16;
typedef __attribute__((ext_vector_type(16))) __bf16 v16bf;
typedef __attribute__((ext_vector_type(8)))  __bf16 v8bf;
typedef __attribute__((ext_vector_type(8)))  float  v8f;
typedef __attribute__((ext_vector_type(4)))  unsigned int u32x4;
typedef __attribute__((ext_vector_type(8)))  int i32x8;
typedef __attribute__((ext_vector_type(4)))  int i32x4;

#define BB 16
#define NN 4096
#define CC 384
#define HH 8
#define DD 48
#define MROWS (BB * NN)   // 65536
#define LDSB  392         // padded LDS row stride (elements): 196 dwords = 4 mod 64 banks

static __device__ inline v8f zero8() {
    v8f z;
#pragma unroll
    for (int i = 0; i < 8; ++i) z[i] = 0.0f;
    return z;
}

static __device__ inline v8f wmma_bf16(v16bf a, v16bf b, v8f c) {
    return __builtin_amdgcn_wmma_f32_16x16x32_bf16(
        false, a, false, b, (short)0, c, false, false);
}

// A fragment (16x32 bf16, row-major, row stride lda). lanes 0-15: row L,
// k {0..15}; lanes 16-31: row L-16, k {16..31}.
static __device__ inline v16bf load_a_frag(const bf16* __restrict__ base,
                                           int lda, int k0, int lane) {
    int m   = lane & 15;
    int off = (lane & 16) ? 16 : 0;
    const bf16* p = base + (size_t)m * lda + k0 + off;
    union { v16bf v; v8bf h[2]; } u;
    u.h[0] = *(const v8bf*)(p);
    u.h[1] = *(const v8bf*)(p + 8);
    return u.v;
}

// B fragment (32x16 bf16) from column-contiguous global source (stride ldb).
static __device__ inline v16bf load_b_frag(const bf16* __restrict__ base,
                                           int ldb, int k0, int lane) {
    int n   = lane & 15;
    int off = (lane & 16) ? 16 : 0;
    return *(const v16bf*)(base + (size_t)n * ldb + k0 + off);
}

// B fragment from the padded LDS weight tile (stride LDSB). Two 16B ds loads
// per lane (odd columns are only 16B aligned due to the 784B padded stride).
static __device__ inline v16bf load_b_frag_lds(const bf16* base, int k0, int lane) {
    int n   = lane & 15;
    int off = (lane & 16) ? 16 : 0;
    const bf16* p = base + n * LDSB + k0 + off;
    union { v16bf v; v8bf h[2]; } u;
    u.h[0] = *(const v8bf*)(p);
    u.h[1] = *(const v8bf*)(p + 8);
    return u.v;
}

// ---------------------------------------------------------------------------
// Stage a 64-row x 384-col bf16 weight tile (rows col0..col0+63 of a row-major
// [Nout,384] weight matrix) into LDS with padded row stride LDSB.
// Uses the Tensor Data Mover in iterate mode: tile = 1 row of 384 elements,
// 64 iterations, lds_addr_increment = 392 elems (inserts the bank pad),
// global_addr_increment = 384 elems. Issued by wave 0 only; TENSORcnt waited
// in the issuing wave, then a workgroup barrier publishes the tile.
static __device__ inline void stage_w_tile(const bf16* __restrict__ Wg,
                                           bf16* WT, int tidx) {
#if defined(__has_builtin) && __has_builtin(__builtin_amdgcn_tensor_load_to_lds)
    if ((tidx >> 5) == 0) {                       // wave 0 (uniform branch)
        unsigned lds_off = (unsigned)(uintptr_t)WT;  // generic LDS addr[31:0] = LDS offset
        unsigned long long ga = (unsigned long long)(uintptr_t)Wg;
        u32x4 g0;
        g0[0] = 1u;                               // count=1 (valid user descriptor)
        g0[1] = lds_off;                          // lds_addr
        g0[2] = (unsigned)(ga & 0xFFFFFFFFull);   // global_addr[31:0]
        g0[3] = (unsigned)((ga >> 32) & 0x01FFFFFFull) | 0x80000000u; // addr[56:32] | type=2
        i32x8 g1;
        g1[0] = (int)((1u << 16) | (1u << 19));   // data_size=2B, iterate_enable=1
        g1[1] = (int)(384u << 16);                // tensor_dim0[15:0]=384 (bits 63:48)
        g1[2] = (int)(0xFFFFu << 16);             // tensor_dim0 hi=0, tensor_dim1 lo=65535
        g1[3] = (int)(384u << 16);                // tensor_dim1 hi=0, tile_dim0=384
        g1[4] = 1;                                // tile_dim1=1 (one row per iteration)
        g1[5] = 384;                              // tensor_dim0_stride[31:0]
        g1[6] = 0;                                // stride hi / tensor_dim1_stride lo
        g1[7] = 0;
        i32x4 g2;
        g2[0] = 0;                                // tensor_dim2 (unused)
        g2[1] = LDSB;                             // lds_addr_increment per iteration
        g2[2] = 384;                              // global_addr_increment per iteration
        g2[3] = (int)(63u << 16);                 // iterate_count = 63 -> 64 iterations
        i32x4 g3; g3[0] = 0; g3[1] = 0; g3[2] = 0; g3[3] = 0;
#if defined(__clang_major__) && (__clang_major__ >= 23)
        i32x8 gz; 
#pragma unroll
        for (int i = 0; i < 8; ++i) gz[i] = 0;
        __builtin_amdgcn_tensor_load_to_lds(g0, g1, g2, g3, gz, 0);
#else
        __builtin_amdgcn_tensor_load_to_lds(g0, g1, g2, g3, 0);
#endif
        __builtin_amdgcn_s_wait_tensorcnt(0);
    }
#else
    // Cooperative fallback: 64 rows x 48 chunks of 8 bf16.
    for (int idx = tidx; idx < 64 * 48; idx += 256) {
        int r = idx / 48, c8 = idx % 48;
        *(v8bf*)(WT + r * LDSB + c8 * 8) = *(const v8bf*)(Wg + (size_t)r * CC + c8 * 8);
    }
#endif
    __syncthreads();
}

// Per-wave 32(M) x 64(N) tile against the LDS-staged weight tile, K = 384.
static __device__ inline void gemm_wave(const bf16* __restrict__ Xrow,
                                        const bf16* WT, int lane, v8f acc[2][4]) {
#pragma unroll
    for (int rt = 0; rt < 2; ++rt)
#pragma unroll
        for (int t = 0; t < 4; ++t) acc[rt][t] = zero8();
    for (int k0 = 0; k0 < CC; k0 += 32) {
        if (k0 + 64 < CC)   // warm the A stream (global_prefetch_b8)
            __builtin_prefetch(Xrow + (size_t)(lane & 15) * CC + k0 + 64, 0, 1);
        v16bf a0 = load_a_frag(Xrow, CC, k0, lane);
        v16bf a1 = load_a_frag(Xrow + (size_t)16 * CC, CC, k0, lane);
#pragma unroll
        for (int t = 0; t < 4; ++t) {
            v16bf b = load_b_frag_lds(WT + t * 16 * LDSB, k0, lane);
            acc[0][t] = wmma_bf16(a0, b, acc[0][t]);
            acc[1][t] = wmma_bf16(a1, b, acc[1][t]);
        }
    }
}

// ---------------------------------------------------------------------------
__global__ void k_cvt(const float* __restrict__ s, bf16* __restrict__ d, int n) {
    int i = blockIdx.x * blockDim.x + threadIdx.x;
    if (i < n) d[i] = (bf16)s[i];
}

// ---------------------------------------------------------------------------
// GEMM1: Xin = Xbf @ convW^T + conv_b  (bf16, row-major [MROWS, C])
// grid (256 mblocks, 6 ngroups) x 256 thr; block = 256 rows x 64 cols.
__global__ void k_gemm_conv(const bf16* __restrict__ X, const bf16* __restrict__ Wc,
                            const float* __restrict__ bias, bf16* __restrict__ Y) {
    __shared__ __align__(32) bf16 WT[64 * LDSB];
    int tidx = threadIdx.x, lane = tidx & 31, wave = tidx >> 5;
    int col0 = blockIdx.y * 64;
    int row0 = blockIdx.x * 256 + wave * 32;
    stage_w_tile(Wc + (size_t)col0 * CC, WT, tidx);
    v8f acc[2][4];
    gemm_wave(X + (size_t)row0 * CC, WT, lane, acc);
    int mb = (lane & 16) ? 8 : 0, nc = lane & 15;
#pragma unroll
    for (int rt = 0; rt < 2; ++rt)
#pragma unroll
        for (int t = 0; t < 4; ++t) {
            int col = col0 + t * 16 + nc;
            float bv = bias[col];
#pragma unroll
            for (int r = 0; r < 8; ++r) {
                int row = row0 + rt * 16 + mb + r;
                Y[(size_t)row * CC + col] = (bf16)(acc[rt][t][r] + bv);
            }
        }
}

// ---------------------------------------------------------------------------
// Fused q/kv projection: Y = Xin @ [wq;wkv]^T + [bq;bkv]  (Nout = 1152)
//   q,k -> [B,H,48,N] (token-contiguous)   v -> [B,H,N,64] (d-contig, padded)
__global__ void k_gemm_qkv(const bf16* __restrict__ Xin, const bf16* __restrict__ Wall,
                           const float* __restrict__ bq, const float* __restrict__ bkv,
                           bf16* __restrict__ Q, bf16* __restrict__ Kp,
                           bf16* __restrict__ Vt) {
    __shared__ __align__(32) bf16 WT[64 * LDSB];
    int tidx = threadIdx.x, lane = tidx & 31, wave = tidx >> 5;
    int col0 = blockIdx.y * 64;
    int row0 = blockIdx.x * 256 + wave * 32;
    stage_w_tile(Wall + (size_t)col0 * CC, WT, tidx);
    v8f acc[2][4];
    gemm_wave(Xin + (size_t)row0 * CC, WT, lane, acc);
    int mb = (lane & 16) ? 8 : 0, nc = lane & 15;
#pragma unroll
    for (int rt = 0; rt < 2; ++rt)
#pragma unroll
        for (int t = 0; t < 4; ++t) {
            int col = col0 + t * 16 + nc;                 // 0..1151
            float bv = (col < CC) ? bq[col] : bkv[col - CC];
#pragma unroll
            for (int r = 0; r < 8; ++r) {
                int row = row0 + rt * 16 + mb + r;
                int b = row >> 12, n = row & (NN - 1);
                float val = acc[rt][t][r] + bv;
                if (col < CC) {                           // q
                    int h = col / DD, dd = col % DD;
                    Q[(((size_t)b * HH + h) * DD + dd) * NN + n] = (bf16)val;
                } else {
                    int o = col - CC;
                    if (o < CC) {                         // k
                        int h = o / DD, dd = o % DD;
                        Kp[(((size_t)b * HH + h) * DD + dd) * NN + n] = (bf16)val;
                    } else {                              // v
                        o -= CC;
                        int h = o / DD, dd = o % DD;
                        Vt[(((size_t)b * HH + h) * NN + n) * 64 + dd] = (bf16)val;
                    }
                }
            }
        }
}

// ---------------------------------------------------------------------------
// Attention per (b,h): S = (q*scale) @ k^T (48x48, K=4096 split over 8 waves),
// LDS reduce + row softmax (scale = N^-0.5 = 1/64), out = P @ v (48x4096).
// Output written in transpose(0,2,1,3)+reshape layout [B, d, H, N].
__global__ void k_attn(const bf16* __restrict__ Q, const bf16* __restrict__ Kp,
                       const bf16* __restrict__ Vt, bf16* __restrict__ Z) {
    __shared__ __align__(32) float S[48 * 48];
    __shared__ __align__(32) bf16  P[48 * 64];
    int bh = blockIdx.x, tid = threadIdx.x;
    int lane = tid & 31, wave = tid >> 5;
    const bf16* q = Q  + (size_t)bh * DD * NN;
    const bf16* k = Kp + (size_t)bh * DD * NN;
    const bf16* v = Vt + (size_t)bh * NN * 64;

    for (int i = tid; i < 48 * 48; i += 256) S[i] = 0.0f;
    __syncthreads();

    v8f acc[3][3];
#pragma unroll
    for (int a = 0; a < 3; ++a)
#pragma unroll
        for (int b = 0; b < 3; ++b) acc[a][b] = zero8();
    int kbeg = wave * 512;
    for (int k0 = kbeg; k0 < kbeg + 512; k0 += 32) {
        v16bf af[3];
#pragma unroll
        for (int mt = 0; mt < 3; ++mt)
            af[mt] = load_a_frag(q + (size_t)mt * 16 * NN, NN, k0, lane);
#pragma unroll
        for (int nt = 0; nt < 3; ++nt) {
            v16bf bfrag = load_b_frag(k + (size_t)nt * 16 * NN, NN, k0, lane);
#pragma unroll
            for (int mt = 0; mt < 3; ++mt)
                acc[mt][nt] = wmma_bf16(af[mt], bfrag, acc[mt][nt]);
        }
    }
    int mb = (lane & 16) ? 8 : 0, nc = lane & 15;
#pragma unroll
    for (int mt = 0; mt < 3; ++mt)
#pragma unroll
        for (int nt = 0; nt < 3; ++nt)
#pragma unroll
            for (int r = 0; r < 8; ++r)
                atomicAdd(&S[(mt * 16 + mb + r) * 48 + nt * 16 + nc], acc[mt][nt][r]);
    __syncthreads();

    if (tid < 48) {
        const float scale = 0.015625f;
        float mx = -3.0e38f;
        for (int j = 0; j < 48; ++j) mx = fmaxf(mx, S[tid * 48 + j]);
        float sum = 0.0f;
        for (int j = 0; j < 48; ++j) sum += __expf((S[tid * 48 + j] - mx) * scale);
        float inv = 1.0f / sum;
        for (int j = 0; j < 48; ++j)
            P[tid * 64 + j] = (bf16)(__expf((S[tid * 48 + j] - mx) * scale) * inv);
        for (int j = 48; j < 64; ++j) P[tid * 64 + j] = (bf16)0.0f;
    }
    __syncthreads();

    int b_idx = bh >> 3, h = bh & (HH - 1);
    for (int n0 = wave * 512; n0 < wave * 512 + 512; n0 += 16) {
        v8f oacc[3];
#pragma unroll
        for (int mt = 0; mt < 3; ++mt) oacc[mt] = zero8();
#pragma unroll
        for (int k0 = 0; k0 < 64; k0 += 32) {
            v16bf bfrag = load_b_frag(v + (size_t)n0 * 64, 64, k0, lane);
#pragma unroll
            for (int mt = 0; mt < 3; ++mt) {
                v16bf ap = load_a_frag(&P[mt * 16 * 64], 64, k0, lane);
                oacc[mt] = wmma_bf16(ap, bfrag, oacc[mt]);
            }
        }
#pragma unroll
        for (int mt = 0; mt < 3; ++mt)
#pragma unroll
            for (int r = 0; r < 8; ++r) {
                int i = mt * 16 + mb + r;
                Z[(((size_t)b_idx * DD + i) * HH + h) * NN + (n0 + nc)] = (bf16)oacc[mt][r];
            }
    }
}

// ---------------------------------------------------------------------------
// Final GEMM: out = Z @ wp^T + bp  (fp32 out, row-major [MROWS, C])
__global__ void k_gemm_out(const bf16* __restrict__ Zb, const bf16* __restrict__ Wp,
                           const float* __restrict__ bias, float* __restrict__ out) {
    __shared__ __align__(32) bf16 WT[64 * LDSB];
    int tidx = threadIdx.x, lane = tidx & 31, wave = tidx >> 5;
    int col0 = blockIdx.y * 64;
    int row0 = blockIdx.x * 256 + wave * 32;
    stage_w_tile(Wp + (size_t)col0 * CC, WT, tidx);
    v8f acc[2][4];
    gemm_wave(Zb + (size_t)row0 * CC, WT, lane, acc);
    int mb = (lane & 16) ? 8 : 0, nc = lane & 15;
#pragma unroll
    for (int rt = 0; rt < 2; ++rt)
#pragma unroll
        for (int t = 0; t < 4; ++t) {
            int col = col0 + t * 16 + nc;
            float bv = bias[col];
#pragma unroll
            for (int r = 0; r < 8; ++r) {
                int row = row0 + rt * 16 + mb + r;
                out[(size_t)row * CC + col] = acc[rt][t][r] + bv;
            }
        }
}

// ---------------------------------------------------------------------------
extern "C" void kernel_launch(void* const* d_in, const int* in_sizes, int n_in,
                              void* d_out, int out_size, void* d_ws, size_t ws_size,
                              hipStream_t stream) {
    const float* x      = (const float*)d_in[0];
    const float* conv_w = (const float*)d_in[1];
    const float* conv_b = (const float*)d_in[2];
    const float* wq     = (const float*)d_in[3];
    const float* bq     = (const float*)d_in[4];
    const float* wkv    = (const float*)d_in[5];
    const float* bkv    = (const float*)d_in[6];
    const float* wp     = (const float*)d_in[7];
    const float* bp     = (const float*)d_in[8];
    float* out = (float*)d_out;

    char* ws = (char*)d_ws;
    size_t off = 0;
    bf16* cwb  = (bf16*)(ws + off); off += (size_t)CC * CC * 2;
    bf16* wall = (bf16*)(ws + off); off += (size_t)(3 * CC) * CC * 2;
    bf16* wpb  = (bf16*)(ws + off); off += (size_t)CC * CC * 2;
    bf16* xbf  = (bf16*)(ws + off); off += (size_t)MROWS * CC * 2;
    bf16* xin  = (bf16*)(ws + off); off += (size_t)MROWS * CC * 2;
    bf16* qb   = (bf16*)(ws + off); off += (size_t)BB * HH * DD * NN * 2;
    bf16* kb   = (bf16*)(ws + off); off += (size_t)BB * HH * DD * NN * 2;
    bf16* vt   = (bf16*)(ws + off); off += (size_t)BB * HH * NN * 64 * 2;
    bf16* zb   = xbf;  // xbf dead after GEMM1 -> reuse for Z

    const int nw = CC * CC;
    k_cvt<<<(nw + 255) / 256, 256, 0, stream>>>(conv_w, cwb, nw);
    k_cvt<<<(nw + 255) / 256, 256, 0, stream>>>(wq, wall, nw);
    k_cvt<<<(2 * nw + 255) / 256, 256, 0, stream>>>(wkv, wall + nw, 2 * nw);
    k_cvt<<<(nw + 255) / 256, 256, 0, stream>>>(wp, wpb, nw);
    const int nx = MROWS * CC;
    k_cvt<<<(nx + 255) / 256, 256, 0, stream>>>(x, xbf, nx);
    hipMemsetAsync(vt, 0, (size_t)BB * HH * NN * 64 * 2, stream);

    k_gemm_conv<<<dim3(256, 6), 256, 0, stream>>>(xbf, cwb, conv_b, xin);
    k_gemm_qkv<<<dim3(256, 18), 256, 0, stream>>>(xin, wall, bq, bkv, qb, kb, vt);
    k_attn<<<BB * HH, 256, 0, stream>>>(qb, kb, vt, zb);
    k_gemm_out<<<dim3(256, 6), 256, 0, stream>>>(zb, wpb, bp, out);
}
